// ResidualVectorQuantizer_16183436771542
// MI455X (gfx1250) — compile-verified, matrix operand order checked by reading, not run on hardware
//
#include <hip/hip_runtime.h>

typedef _Float16 v16h __attribute__((ext_vector_type(16)));
typedef _Float16 v8h  __attribute__((ext_vector_type(8)));
typedef _Float16 v4h  __attribute__((ext_vector_type(4)));
typedef float    v8f  __attribute__((ext_vector_type(8)));

#define NUM_CB   8
#define KCODES   1024
#define DIM      512
#define NVEC     65536
#define ROWS_PB  128   // 8 waves * 16 rows
#define NWAVES   8

union AFrag { v16h v; _Float16 e[16]; };
union BFrag { v16h v; v8h h[2]; };

// ---------------------------------------------------------------------------
// Prep: codebooks f32 -> f16, and cnorm[k] = sum_d cb[k][d]^2
// One block per codebook row (8192 blocks, 128 threads).
// ---------------------------------------------------------------------------
__global__ void rvq_prep_kernel(const float* __restrict__ cb,
                                _Float16* __restrict__ cbh,
                                float* __restrict__ cnorm) {
    __shared__ float red[128];
    const int rowId = blockIdx.x;            // 0 .. NUM_CB*KCODES-1
    const int tid   = threadIdx.x;           // 0 .. 127
    const float* rp = cb + (size_t)rowId * DIM;
    _Float16*    hp = cbh + (size_t)rowId * DIM;

    const int d0 = tid * 4;
    float4 v = *(const float4*)(rp + d0);
    v4h hv;
    hv.x = (_Float16)v.x; hv.y = (_Float16)v.y;
    hv.z = (_Float16)v.z; hv.w = (_Float16)v.w;
    *(v4h*)(hp + d0) = hv;

    float ss = v.x * v.x + v.y * v.y + v.z * v.z + v.w * v.w;
    red[tid] = ss;
    __syncthreads();
    #pragma unroll
    for (int off = 64; off >= 1; off >>= 1) {
        if (tid < off) red[tid] += red[tid + off];
        __syncthreads();
    }
    if (tid == 0) cnorm[rowId] = red[0];
}

// ---------------------------------------------------------------------------
// One RVQ step: distances via v_wmma_f32_16x16x32_f16, argmin, residual update.
// grid = NVEC/ROWS_PB = 512 blocks, 256 threads (8 waves, 16 rows per wave).
// src: residual input (x for step 0, resbuf otherwise). resbuf updated in place.
// ---------------------------------------------------------------------------
__global__ void __launch_bounds__(256)
rvq_step_kernel(const float* __restrict__ src,          // [NVEC, DIM] residual in
                float* __restrict__ resbuf,             // [NVEC, DIM] residual out
                const _Float16* __restrict__ cbh_s,     // [KCODES, DIM] f16 codebook slice
                const float* __restrict__ cbf_s,        // [KCODES, DIM] f32 codebook slice
                const float* __restrict__ cnorm_s,      // [KCODES]
                const float* __restrict__ x,            // [NVEC, DIM] original input
                float* __restrict__ outSum,             // [NVEC, DIM] (written iff last)
                float* __restrict__ outIdx,             // [NVEC, NUM_CB] (as float)
                int step, int isLast) {
    __shared__ int sIdx[NWAVES * 16];

    const int tid   = threadIdx.x;
    const int lane  = tid & 31;              // wave32
    const int wid   = tid >> 5;
    const int nlane = lane & 15;             // column-within-tile this lane owns
    const int half  = (lane < 16) ? 0 : 8;   // K-chunk half per WMMA A/B layout
    const int rowBase = blockIdx.x * ROWS_PB + wid * 16;

    // ---- Load this wave's 16 rows as f16 A fragments for all D=512 ----------
    // Chunk c covers d in [32c, 32c+32). Lane layout (16-bit A 16x32):
    //   lanes 0-15 : K = {0..7, 16..23}; lanes 16-31 : K = {8..15, 24..31}
    const float* arow = src + (size_t)(rowBase + nlane) * DIM;
    v16h a[16];
    #pragma unroll
    for (int c = 0; c < 16; ++c) {
        const int d0 = c * 32 + half;
        float4 f0 = *(const float4*)(arow + d0);
        float4 f1 = *(const float4*)(arow + d0 + 4);
        float4 f2 = *(const float4*)(arow + d0 + 16);
        float4 f3 = *(const float4*)(arow + d0 + 20);
        AFrag af;
        af.e[0]  = (_Float16)f0.x; af.e[1]  = (_Float16)f0.y;
        af.e[2]  = (_Float16)f0.z; af.e[3]  = (_Float16)f0.w;
        af.e[4]  = (_Float16)f1.x; af.e[5]  = (_Float16)f1.y;
        af.e[6]  = (_Float16)f1.z; af.e[7]  = (_Float16)f1.w;
        af.e[8]  = (_Float16)f2.x; af.e[9]  = (_Float16)f2.y;
        af.e[10] = (_Float16)f2.z; af.e[11] = (_Float16)f2.w;
        af.e[12] = (_Float16)f3.x; af.e[13] = (_Float16)f3.y;
        af.e[14] = (_Float16)f3.z; af.e[15] = (_Float16)f3.w;
        a[c] = af.v;
    }

    // ---- Sweep all 1024 codes in 16-wide column tiles ----------------------
    float minv[8];
    int   mini[8];
    #pragma unroll
    for (int j = 0; j < 8; ++j) { minv[j] = 3.4e38f; mini[j] = 0; }

    for (int k0 = 0; k0 < KCODES; k0 += 16) {
        const int col = k0 + nlane;
        const _Float16* brow = cbh_s + (size_t)col * DIM + half;
        v8f acc = {};
        #pragma unroll
        for (int c = 0; c < 16; ++c) {
            BFrag bf;
            bf.h[0] = *(const v8h*)(brow + c * 32);
            bf.h[1] = *(const v8h*)(brow + c * 32 + 16);
            acc = __builtin_amdgcn_wmma_f32_16x16x32_f16(
                false, a[c], false, bf.v, (short)0, acc, false, false);
        }
        const float cn = cnorm_s[col];
        #pragma unroll
        for (int j = 0; j < 8; ++j) {
            // ||r||^2 term is row-constant: drop it from the argmin.
            float d2 = cn - 2.0f * acc[j];
            if (d2 < minv[j]) { minv[j] = d2; mini[j] = col; }
        }
    }

    // ---- Cross-lane argmin: C layout keeps rows within 16-lane halves ------
    #pragma unroll
    for (int j = 0; j < 8; ++j) {
        #pragma unroll
        for (int off = 8; off >= 1; off >>= 1) {
            float ov = __shfl_xor(minv[j], off, 32);
            int   oi = __shfl_xor(mini[j], off, 32);
            if (ov < minv[j] || (ov == minv[j] && oi < mini[j])) {
                minv[j] = ov; mini[j] = oi;
            }
        }
    }
    // lanes 0..15 now agree on rows m=0..7 (slot j=m); lanes 16..31 on m=8..15
    if (lane == 0 || lane == 16) {
        const int mbase = (lane == 0) ? 0 : 8;
        #pragma unroll
        for (int j = 0; j < 8; ++j) {
            sIdx[wid * 16 + mbase + j] = mini[j];
            outIdx[(size_t)(rowBase + mbase + j) * NUM_CB + step] = (float)mini[j];
        }
    }
    __syncthreads();

    // ---- Residual update (exact f32): r' = r - cb[idx]; out = x - r' -------
    {
        const int rl  = lane >> 1;                       // 0..15
        const int row = rowBase + rl;
        const int bi  = sIdx[wid * 16 + rl];
        const float4* qv = (const float4*)(cbf_s + (size_t)bi * DIM);
        const float4* sv = (const float4*)(src + (size_t)row * DIM);
        const float4* xv = (const float4*)(x + (size_t)row * DIM);
        float4*       rv = (float4*)(resbuf + (size_t)row * DIM);
        float4*       ov = (float4*)(outSum + (size_t)row * DIM);
        const int t0 = (lane & 1) * 64;                  // half a row in float4s
        for (int t = t0; t < t0 + 64; ++t) {
            float4 r = sv[t], q = qv[t];
            float4 nr;
            nr.x = r.x - q.x; nr.y = r.y - q.y;
            nr.z = r.z - q.z; nr.w = r.w - q.w;
            rv[t] = nr;
            if (isLast) {
                float4 xx = xv[t];
                float4 o;
                o.x = xx.x - nr.x; o.y = xx.y - nr.y;
                o.z = xx.z - nr.z; o.w = xx.w - nr.w;
                ov[t] = o;
            }
        }
    }
}

// ---------------------------------------------------------------------------
extern "C" void kernel_launch(void* const* d_in, const int* in_sizes, int n_in,
                              void* d_out, int out_size, void* d_ws, size_t ws_size,
                              hipStream_t stream) {
    const float* x  = (const float*)d_in[0];   // [NVEC, DIM]
    const float* cb = (const float*)d_in[1];   // [NUM_CB, KCODES, DIM]
    float* out      = (float*)d_out;           // [NVEC*DIM] sum ++ [NVEC*NUM_CB] idx

    // Workspace carve
    char* w = (char*)d_ws;
    float* resbuf = (float*)w;                 w += (size_t)NVEC * DIM * sizeof(float);
    _Float16* cbh = (_Float16*)w;              w += (size_t)NUM_CB * KCODES * DIM * sizeof(_Float16);
    float* cnorm  = (float*)w;

    rvq_prep_kernel<<<NUM_CB * KCODES, 128, 0, stream>>>(cb, cbh, cnorm);

    for (int s = 0; s < NUM_CB; ++s) {
        const float* src = (s == 0) ? x : (const float*)resbuf;
        rvq_step_kernel<<<NVEC / ROWS_PB, 256, 0, stream>>>(
            src, resbuf,
            cbh + (size_t)s * KCODES * DIM,
            cb  + (size_t)s * KCODES * DIM,
            cnorm + (size_t)s * KCODES,
            x, out, out + (size_t)NVEC * DIM,
            s, (s == NUM_CB - 1) ? 1 : 0);
    }
}
